// Conv_ImgEncoder_40510131536217
// MI455X (gfx1250) — matrix operand order repaired and checked by reading, not compiled, not run on hardware
//
#include <hip/hip_runtime.h>

typedef __bf16    v16bf __attribute__((ext_vector_type(16)));
typedef __bf16    v2bf  __attribute__((ext_vector_type(2)));
typedef float     v8f   __attribute__((ext_vector_type(8)));
typedef unsigned  v8u   __attribute__((ext_vector_type(8)));
typedef int       v4i   __attribute__((ext_vector_type(4)));

#if __has_builtin(__builtin_amdgcn_global_load_async_to_lds_b128)
#define HAVE_ASYNC_LDS 1
typedef __attribute__((address_space(1))) v4i* gvec_ptr;
typedef __attribute__((address_space(3))) v4i* lvec_ptr;
#endif

// ---------- bf16 helpers (hardware RNE conversion) ----------
__device__ __forceinline__ unsigned pack2bf(float a, float b) {
  v2bf r;
  r[0] = (__bf16)a;
  r[1] = (__bf16)b;
  return __builtin_bit_cast(unsigned, r);
}
__device__ __forceinline__ float lrelu(float v) { return v >= 0.f ? v : 0.2f * v; }

// =====================================================================
// 1) crop/resize per (batch, mask): _trans_img
// =====================================================================
__global__ __launch_bounds__(256) void trans_img_kernel(
    const float* __restrict__ img, const float* __restrict__ mask,
    float* __restrict__ images) {
  const int b = blockIdx.x >> 3;
  const int s = blockIdx.x & 7;
  const int tid = threadIdx.x;
  __shared__ float colsum[256];
  __shared__ float rowsum[256];
  __shared__ int   box[4];   // minR, maxR, minC, maxC

  colsum[tid] = 0.f; rowsum[tid] = 0.f;
  if (tid < 4) box[tid] = (tid & 1) ? -1 : 0x7FFFFFFF;
  __syncthreads();

  const float* mptr = mask + (b * 8 + s) * 65536;
  const float* i0   = img  + b * 3 * 65536;

  for (int idx = tid; idx < 65536; idx += 256) {
    float m = mptr[idx];
    if (m != 0.f) {
      int y = idx >> 8, x = idx & 255;
      float v = (i0[idx] + i0[65536 + idx] + i0[131072 + idx]) * m;
      atomicAdd(&colsum[x], v);
      atomicAdd(&rowsum[y], v);
    }
  }
  __syncthreads();
  if (colsum[tid] != 0.f) { atomicMin(&box[2], tid); atomicMax(&box[3], tid); }
  if (rowsum[tid] != 0.f) { atomicMin(&box[0], tid); atomicMax(&box[1], tid); }
  __syncthreads();

  const int x0 = box[0], x1 = box[1], y0 = box[2], y1 = box[3];
  const bool ok = (x1 >= 0) && (y1 >= 0);
  const int h = x1 - x0 + 1;
  const int w = y1 - y0 + 1;

  float* outp = images + (b * 24 + s * 3) * 65536;
  for (int idx = tid; idx < 65536; idx += 256) {
    int y = idx >> 8, x = idx & 255;
    int ry, cx;
    if (ok) { ry = x0 + ((y * h) >> 8); cx = y0 + ((x * w) >> 8); }
    else    { ry = y; cx = x; }
    int sidx = (ry << 8) | cx;
    float m = mptr[sidx];
    outp[idx]          = i0[sidx]          * m;
    outp[65536 + idx]  = i0[65536 + sidx]  * m;
    outp[131072 + idx] = i0[131072 + sidx] * m;
  }
}

// =====================================================================
// 2) spectral-norm scale: invSigma = 1 / || M @ normalize(M^T u) ||
// =====================================================================
__global__ __launch_bounds__(256) void sigma_kernel(
    const float* __restrict__ w, float* __restrict__ sigout,
    int O, int Kf, unsigned seed) {
  __shared__ float su[2048];
  __shared__ float sv[1184];
  __shared__ float red[256];
  const int tid = threadIdx.x;

  for (int i = tid; i < O; i += 256) {
    unsigned h = (unsigned)i * 2654435761u ^ (seed * 0x9E3779B9u);
    h ^= h >> 13; h *= 0x85EBCA6Bu; h ^= h >> 16;
    su[i] = (float)(h & 0xFFFFFFu) * (1.f / 8388608.f) - 1.f;
  }
  __syncthreads();

  float p = 0.f;
  for (int i = tid; i < O; i += 256) p += su[i] * su[i];
  red[tid] = p; __syncthreads();
  for (int off = 128; off > 0; off >>= 1) {
    if (tid < off) red[tid] += red[tid + off];
    __syncthreads();
  }
  float inn = rsqrtf(red[0] + 1e-24f);
  __syncthreads();
  for (int i = tid; i < O; i += 256) su[i] *= inn;
  __syncthreads();

  for (int j = tid; j < Kf; j += 256) {
    float a = 0.f;
    for (int o = 0; o < O; ++o) a += su[o] * w[o * Kf + j];
    sv[j] = a;
  }
  __syncthreads();
  p = 0.f;
  for (int j = tid; j < Kf; j += 256) p += sv[j] * sv[j];
  red[tid] = p; __syncthreads();
  for (int off = 128; off > 0; off >>= 1) {
    if (tid < off) red[tid] += red[tid + off];
    __syncthreads();
  }
  inn = rsqrtf(red[0] + 1e-24f);
  __syncthreads();
  for (int j = tid; j < Kf; j += 256) sv[j] *= inn;
  __syncthreads();

  p = 0.f;
  for (int o = tid; o < O; o += 256) {
    const float* row = w + o * Kf;
    float a = 0.f;
    for (int j = 0; j < Kf; ++j) a += row[j] * sv[j];
    p += a * a;
  }
  red[tid] = p; __syncthreads();
  for (int off = 128; off > 0; off >>= 1) {
    if (tid < off) red[tid] += red[tid + off];
    __syncthreads();
  }
  if (tid == 0) sigout[0] = rsqrtf(red[0] + 1e-12f);
}

// =====================================================================
// 3) weight prep: f32 -> bf16 pairs, K padded to kp*2, invSigma folded
//    grid: (ceil(kp/256), rows)
// =====================================================================
__global__ __launch_bounds__(256) void prep_weights_kernel(
    const float* __restrict__ w, unsigned* __restrict__ outp,
    const float* __restrict__ sig, int K, int kp) {
  const int p = blockIdx.x * 256 + threadIdx.x;
  if (p >= kp) return;
  const int row = blockIdx.y;
  const float s = sig ? sig[0] : 1.f;
  const int k = p * 2;
  float a0 = (k < K)     ? w[row * K + k]     * s : 0.f;
  float a1 = (k + 1 < K) ? w[row * K + k + 1] * s : 0.f;
  outp[row * kp + p] = pack2bf(a0, a1);
}

// =====================================================================
// 4) grouped 3x3 conv as implicit GEMM on v_wmma_f32_16x16x32_bf16
//    one wave computes a 16(Cout) x 16(pixels) tile, K = CinG*9
// =====================================================================
__global__ __launch_bounds__(32) void conv3x3_wmma_kernel(
    const float* __restrict__ in, const unsigned* __restrict__ wbf,
    float* __restrict__ out,
    int CinG, int CoutG, int Hin, int hs, int ntShift, int stride,
    float slope, int kp) {
  const int lane = threadIdx.x;
  const int g = blockIdx.y;
  const int b = blockIdx.z;
  const int mt = blockIdx.x >> ntShift;
  const int nt = blockIdx.x & ((1 << ntShift) - 1);
  const int Hout = 1 << hs;
  const int HW = Hout << hs;          // Hout*Hout
  const int K = CinG * 9;
  const int kch = kp >> 4;            // kp = kch*16 pairs per row

  const int half = lane >> 4;         // K half select (ISA 16-bit layout)
  const int l15 = lane & 15;
  const int n0 = nt * 16;

  int m = mt * 16 + l15;
  int mc = m < CoutG ? m : CoutG - 1;               // clamp (CoutG==8 case)
  const unsigned* wrow = wbf + (g * CoutG + mc) * kp;
  const int kRow = l15 + half * 16;

  // hoisted im2col geometry for this wave's 16 output pixels
  int iy0[16], ix0[16];
#pragma unroll
  for (int n = 0; n < 16; ++n) {
    int nn = n0 + n;
    iy0[n] = (nn >> hs) * stride - 1;
    ix0[n] = (nn & (Hout - 1)) * stride - 1;
  }

  v8f acc = {};
  for (int kc = 0; kc < kch; ++kc) {
    // ---- A fragment: two 16B loads from padded bf16 weights ----
    const uint4* wchunk = (const uint4*)(wrow + kc * 16 + (half << 2));
    uint4 lo = wchunk[0];
    uint4 hi = wchunk[2];
    v8u ap;
    ap[0] = lo.x; ap[1] = lo.y; ap[2] = lo.z; ap[3] = lo.w;
    ap[4] = hi.x; ap[5] = hi.y; ap[6] = hi.z; ap[7] = hi.w;

    // ---- B fragment: im2col row kB, clamped addresses (no branches) ----
    const int nn = n0 + l15;
    const int oy = (nn >> hs) * stride - 1;
    const int ox = (nn & (Hout - 1)) * stride - 1;
    const float* ipb = in + ((b * 8 + g) * CinG) * Hin * Hin;
    float bv[16];
#pragma unroll
    for (int n = 0; n < 16; ++n) {
      const int kB = kc * 32 + (half << 3) + ((n & 8) << 1) + (n & 7);
      const bool kOK = kB < K;
      const int kBc = kOK ? kB : K - 1;
      const int ci = kBc / 9;
      const int rem = kBc - ci * 9;
      const int ky = rem / 3;
      const int kx = rem - ky * 3;
      int iy = oy + ky;
      int ix = ox + kx;
      bool inb = ((unsigned)iy < (unsigned)Hin) & ((unsigned)ix < (unsigned)Hin);
      int off = inb ? (iy * Hin + ix) : 0;          // clamped address
      float v = ipb[ci * Hin * Hin + off];          // unconditional load
      v = (inb & kOK) ? v : 0.f;                    // cndmask
      bv[n] = v >= 0.f ? v : v * slope;             // fused input act
    }
    v8u bp;
#pragma unroll
    for (int j = 0; j < 8; ++j) bp[j] = pack2bf(bv[2 * j], bv[2 * j + 1]);

    acc = __builtin_amdgcn_wmma_f32_16x16x32_bf16(
        false, __builtin_bit_cast(v16bf, ap),
        false, __builtin_bit_cast(v16bf, bp),
        (short)0, acc, false, false);
  }

  // epilogue: 32-bit addressing, branch-free when the M-tile is full
  const int obase = ((b * 8 + g) * CoutG + mt * 16 + half * 8) * HW + n0 + l15;
  if (CoutG >= 16) {
#pragma unroll
    for (int r = 0; r < 8; ++r) out[obase + r * HW] = acc[r];
  } else {
#pragma unroll
    for (int r = 0; r < 8; ++r) {
      int mm = mt * 16 + r + half * 8;
      if (mm < CoutG) out[obase + r * HW] = acc[r];
    }
  }
}

// =====================================================================
// 5) instance norm (optionally fused lrelu), block per (b,c) plane
// =====================================================================
__global__ __launch_bounds__(256) void inorm_kernel(
    const float* __restrict__ src, float* __restrict__ dst, int HW, int act) {
  __shared__ float red[256];
  __shared__ float red2[256];
  const int base = blockIdx.x * HW;
  const int tid = threadIdx.x;
  float s = 0.f, s2 = 0.f;
  for (int i = tid; i < HW; i += 256) {
    float v = src[base + i];
    s += v; s2 += v * v;
  }
  red[tid] = s; red2[tid] = s2; __syncthreads();
  for (int off = 128; off > 0; off >>= 1) {
    if (tid < off) { red[tid] += red[tid + off]; red2[tid] += red2[tid + off]; }
    __syncthreads();
  }
  const float invHW = 1.f / (float)HW;
  const float mean = red[0] * invHW;
  const float var = red2[0] * invHW - mean * mean;
  const float rs = rsqrtf(var + 1e-5f);
  for (int i = tid; i < HW; i += 256) {
    float v = (src[base + i] - mean) * rs;
    if (act) v = lrelu(v);
    dst[base + i] = v;
  }
}

// =====================================================================
// 6) 1x1-conv heads: 4-wave blocks, bf16 A slab staged to LDS with
//    async global->LDS copies (ASYNCcnt) when available
// =====================================================================
__global__ __launch_bounds__(128) void gemm1x1_wmma_kernel(
    const unsigned* __restrict__ Abf, const float* __restrict__ Bx,
    const float* __restrict__ bias, float* __restrict__ out,
    int M, int N, int K, int kp) {
  __shared__ unsigned ldsA[16 * 16];   // 16 rows x 32 k as bf16 pairs
  const int tid = threadIdx.x;
  const int lane = tid & 31;
  const int wv = tid >> 5;
  const int mt = blockIdx.x;
  const int nbase = blockIdx.y * 64 + wv * 16;
  const int b = blockIdx.z;
  const int half = lane >> 4;
  const int l15 = lane & 15;

  const float* Bbatch = Bx + b * K * N;
  v8f acc = {};
  const int kch = K >> 5;

  for (int kc = 0; kc < kch; ++kc) {
    if (wv == 0) {
#if defined(HAVE_ASYNC_LDS)
      // 64 x 16B segments; wave 0 issues 2 async b128 copies per lane
#pragma unroll
      for (int it = 0; it < 2; ++it) {
        int seg = lane + it * 32;            // 0..63
        int row = seg >> 2, part = seg & 3;
        const unsigned* gsrc = Abf + (mt * 16 + row) * kp + kc * 16 + part * 4;
        __builtin_amdgcn_global_load_async_to_lds_b128(
            (gvec_ptr)gsrc, (lvec_ptr)&ldsA[seg * 4], 0, 0);
      }
#if __has_builtin(__builtin_amdgcn_s_wait_asynccnt)
      __builtin_amdgcn_s_wait_asynccnt(0);
#else
      asm volatile("s_wait_asynccnt 0x0" ::: "memory");
#endif
#else
#pragma unroll
      for (int it = 0; it < 8; ++it) {
        int e = lane + it * 32;
        int row = e >> 4, cu = e & 15;
        ldsA[e] = Abf[(mt * 16 + row) * kp + kc * 16 + cu];
      }
#endif
    }
    __syncthreads();

    v8u ap;
#pragma unroll
    for (int r = 0; r < 8; ++r) {
      int cu = (half << 2) + ((r & 4) << 1) + (r & 3);
      ap[r] = ldsA[l15 * 16 + cu];
    }

    int kB = kc * 32 + l15 + half * 16;
    const float* brow = Bbatch + (kc * 32) * N + nbase + l15;
    float bv[16];
#pragma unroll
    for (int q = 0; q < 16; ++q) {
      int kk = (half << 3) + ((q & 8) << 1) + (q & 7);
      bv[q] = lrelu(brow[kk * N]);
    }
    v8u bp;
#pragma unroll
    for (int j = 0; j < 8; ++j) bp[j] = pack2bf(bv[2 * j], bv[2 * j + 1]);

    if (kc + 1 < kch)
      __builtin_prefetch(Bbatch + (kB + 32) * N + nbase, 0, 1);

    acc = __builtin_amdgcn_wmma_f32_16x16x32_bf16(
        false, __builtin_bit_cast(v16bf, ap),
        false, __builtin_bit_cast(v16bf, bp),
        (short)0, acc, false, false);
    __syncthreads();
  }

  // epilogue: vector bias load, branch-free 32-bit stores
  const int mrow = mt * 16 + half * 8;
  float4 b0 = *reinterpret_cast<const float4*>(bias + mrow);
  float4 b1 = *reinterpret_cast<const float4*>(bias + mrow + 4);
  float bb[8] = {b0.x, b0.y, b0.z, b0.w, b1.x, b1.y, b1.z, b1.w};
  const int obase = (b * M + mrow) * N + nbase + l15;
#pragma unroll
  for (int r = 0; r < 8; ++r) {
    float v = lrelu(acc[r] + bb[r]);
    out[obase + r * N] = v;
  }
}

// =====================================================================
// launch
// =====================================================================
extern "C" void kernel_launch(void* const* d_in, const int* in_sizes, int n_in,
                              void* d_out, int out_size, void* d_ws, size_t ws_size,
                              hipStream_t stream) {
  (void)in_sizes; (void)n_in; (void)out_size; (void)ws_size;
  const float* real = (const float*)d_in[0];
  const float* sem  = (const float*)d_in[1];
  const float* w1 = (const float*)d_in[2];
  const float* w2 = (const float*)d_in[3];
  const float* w3 = (const float*)d_in[4];
  const float* w4 = (const float*)d_in[5];
  const float* w5 = (const float*)d_in[6];
  const float* w6 = (const float*)d_in[7];
  const float* gw1 = (const float*)d_in[8];
  const float* gb1 = (const float*)d_in[9];
  const float* gw2 = (const float*)d_in[10];
  const float* gb2 = (const float*)d_in[11];

  float* outX   = (float*)d_out;                 // (8,2048,8,8)
  float* outX1  = outX + 1048576;                // (8,2048,16,16)
  float* outX2  = outX1 + 4194304;               // (8,2048,32,32)
  float* images = outX2 + 16777216;              // (8,24,256,256)

  // big intermediates parked inside the x_2 region (written last)
  float* t1 = outX2;                             // (8,64,128,128)  8388608
  float* t2 = outX2 + 8388608;                   // (8,128,64,64)   4194304

  float* ws  = (float*)d_ws;
  float* sig = ws;                               // 16 floats
  float* t8  = ws + 16;                          // (8,256,32,32)   2097152
  float* t16 = t8 + 2097152;                     // (8,512,16,16)   1048576
  float* t32 = t16 + 1048576;                    // (8,1024,8,8)     524288
  float* t6  = t32 + 524288;                     // (8,2048,8,8)    1048576

  // padded bf16 weights (uint = bf16 pair); 16B-aligned region
  unsigned* wb1  = (unsigned*)(t6 + 1048576);    //   64 x  16
  unsigned* wb2  = wb1 + 64 * 16;                //  128 x  48
  unsigned* wb3  = wb2 + 128 * 48;               //  256 x  80
  unsigned* wb4  = wb3 + 256 * 80;               //  512 x 144
  unsigned* wb5  = wb4 + 512 * 144;              // 1024 x 288
  unsigned* wb6  = wb5 + 1024 * 288;             // 2048 x 576
  unsigned* wbg1 = wb6 + 2048 * 576;             // 2048 x 256
  unsigned* wbg2 = wbg1 + 2048 * 256;            // 2048 x 128

  trans_img_kernel<<<64, 256, 0, stream>>>(real, sem, images);

  sigma_kernel<<<1, 256, 0, stream>>>(w1, sig + 0,   64,   27, 1u);
  sigma_kernel<<<1, 256, 0, stream>>>(w2, sig + 1,  128,   72, 2u);
  sigma_kernel<<<1, 256, 0, stream>>>(w3, sig + 2,  256,  144, 3u);
  sigma_kernel<<<1, 256, 0, stream>>>(w4, sig + 3,  512,  288, 4u);
  sigma_kernel<<<1, 256, 0, stream>>>(w5, sig + 4, 1024,  576, 5u);
  sigma_kernel<<<1, 256, 0, stream>>>(w6, sig + 5, 2048, 1152, 6u);

  prep_weights_kernel<<<dim3(1,   64), 256, 0, stream>>>(w1,  wb1,  sig + 0,   27,  16);
  prep_weights_kernel<<<dim3(1,  128), 256, 0, stream>>>(w2,  wb2,  sig + 1,   72,  48);
  prep_weights_kernel<<<dim3(1,  256), 256, 0, stream>>>(w3,  wb3,  sig + 2,  144,  80);
  prep_weights_kernel<<<dim3(1,  512), 256, 0, stream>>>(w4,  wb4,  sig + 3,  288, 144);
  prep_weights_kernel<<<dim3(2, 1024), 256, 0, stream>>>(w5,  wb5,  sig + 4,  576, 288);
  prep_weights_kernel<<<dim3(3, 2048), 256, 0, stream>>>(w6,  wb6,  sig + 5, 1152, 576);
  prep_weights_kernel<<<dim3(1, 2048), 256, 0, stream>>>(gw1, wbg1, nullptr,  512, 256);
  prep_weights_kernel<<<dim3(1, 2048), 256, 0, stream>>>(gw2, wbg2, nullptr,  256, 128);

  // layer 1: images(24ch,256) -> t1(64ch,128)   CinG=3 CoutG=8
  conv3x3_wmma_kernel<<<dim3(1024, 8, 8), 32, 0, stream>>>(
      images, wb1, t1, 3, 8, 256, 7, 10, 2, 1.0f, 16);
  inorm_kernel<<<512, 256, 0, stream>>>(t1, t1, 16384, 0);

  // layer 2: t1 -> t2(128ch,64)                CinG=8 CoutG=16
  conv3x3_wmma_kernel<<<dim3(256, 8, 8), 32, 0, stream>>>(
      t1, wb2, t2, 8, 16, 128, 6, 8, 2, 0.2f, 48);
  inorm_kernel<<<1024, 256, 0, stream>>>(t2, t2, 4096, 0);

  // layer 3: t2 -> t8(256ch,32)                CinG=16 CoutG=32
  conv3x3_wmma_kernel<<<dim3(128, 8, 8), 32, 0, stream>>>(
      t2, wb3, t8, 16, 32, 64, 5, 6, 2, 0.2f, 80);
  inorm_kernel<<<2048, 256, 0, stream>>>(t8, t8, 1024, 0);

  // layer 4: t8 -> t16(512ch,16)               CinG=32 CoutG=64
  conv3x3_wmma_kernel<<<dim3(64, 8, 8), 32, 0, stream>>>(
      t8, wb4, t16, 32, 64, 32, 4, 4, 2, 0.2f, 144);
  inorm_kernel<<<4096, 256, 0, stream>>>(t16, t16, 256, 0);

  // layer 5: t16 -> t32(1024ch,8)              CinG=64 CoutG=128
  conv3x3_wmma_kernel<<<dim3(32, 8, 8), 32, 0, stream>>>(
      t16, wb5, t32, 64, 128, 16, 3, 2, 2, 0.2f, 288);
  inorm_kernel<<<8192, 256, 0, stream>>>(t32, t32, 64, 0);

  // layer 6: t32 -> t6(2048ch,8), stride 1     CinG=128 CoutG=256
  conv3x3_wmma_kernel<<<dim3(64, 8, 8), 32, 0, stream>>>(
      t32, wb6, t6, 128, 256, 8, 3, 2, 1, 0.2f, 576);
  // x = lrelu(inorm(t6)) -> d_out
  inorm_kernel<<<16384, 256, 0, stream>>>(t6, outX, 64, 1);

  // head 1: x_1 = lrelu(gw1 @ lrelu(t16) + gb1)   M=2048 N=256 K=512
  gemm1x1_wmma_kernel<<<dim3(128, 4, 8), 128, 0, stream>>>(
      wbg1, t16, gb1, outX1, 2048, 256, 512, 256);

  // head 2: x_2 = lrelu(gw2 @ lrelu(t8) + gb2)    M=2048 N=1024 K=256
  gemm1x1_wmma_kernel<<<dim3(128, 16, 8), 128, 0, stream>>>(
      wbg2, t8, gb2, outX2, 2048, 1024, 256, 128);
}
